// TLGAttention_23441931501666
// MI455X (gfx1250) — matrix operand order repaired and checked by reading, not compiled, not run on hardware
//
#include <hip/hip_runtime.h>
#include <hip/hip_bf16.h>
#include <cstdint>
#include <cstddef>

// ---------------------------------------------------------------------------
// TLG attention for MI455X (gfx1250), wave32, WMMA bf16->f32.
//   C=128, D=16, H=32, W=32, L=8 (two halves of 4), heads=8, head_dim=16,
//   windows LOCAL=(1,2,2,2) -> 8 positions per window.
// ---------------------------------------------------------------------------

#define C_CH   128
#define DHW    16384      // 16*32*32
#define KTOT   384        // 3 taps * 128 channels

typedef __attribute__((ext_vector_type(16))) __bf16 v16bf;
typedef __attribute__((ext_vector_type(8)))  float  v8f;

union BF16Frag {
    v16bf v;
    uint4 q[2];
    unsigned short s[16];
};

__device__ __forceinline__ unsigned short f2bf(float f) {
    unsigned int u = __float_as_uint(f);
    unsigned int r = (u + 0x7FFFu + ((u >> 16) & 1u)) >> 16;  // RNE
    return (unsigned short)r;
}

// ---------------------------------------------------------------------------
// Kernel 0: transpose+convert weights (C_out, C_in, 3) fp32 -> bf16 [o][K],
//           K = tap*128 + c  (so each B fragment is two runs of 8 contiguous bf16).
// ---------------------------------------------------------------------------
__global__ void wt_transpose_kernel(const float* __restrict__ wq,
                                    const float* __restrict__ wk,
                                    const float* __restrict__ wv,
                                    const float* __restrict__ wp,
                                    unsigned short* __restrict__ Wt) {
    int idx = blockIdx.x * blockDim.x + threadIdx.x;
    if (idx >= 4 * C_CH * KTOT) return;
    int mat = idx / (C_CH * KTOT);
    int rem = idx - mat * (C_CH * KTOT);
    int o = rem / KTOT;
    int k = rem - o * KTOT;
    int t = k / 128;
    int c = k - t * 128;
    const float* src = (mat == 0) ? wq : (mat == 1) ? wk : (mat == 2) ? wv : wp;
    Wt[idx] = f2bf(src[(o * 128 + c) * 3 + t]);
}

// ---------------------------------------------------------------------------
// Kernel 1: fused QKV conv (WMMA GEMM, M=16 positions = 2 windows) + window
//           attention. One 128-thread block (4 waves) per window pair.
//   combo 0: q from half 0 (xs), k/v from half 1 (xe)  -> yse
//   combo 1: q from half 1,      k/v from half 0       -> yes
// ---------------------------------------------------------------------------
__global__ __launch_bounds__(128)
void qkv_attn_kernel(const float* __restrict__ x,
                     const float* __restrict__ bq,
                     const float* __restrict__ bk,
                     const float* __restrict__ bv,
                     const unsigned short* __restrict__ Wt,
                     float* __restrict__ Yattn) {
    // [half][pos 16][K 384] bf16 = 24576 B, later reused as float[3][16][128]
    __shared__ __align__(16) unsigned short smem_u16[2 * 16 * KTOT];
    float* smemF = (float*)smem_u16;

    int id = blockIdx.x;
    int wqi   = id & 7;          // w window-pair index (covers w = 4*wqi..+3)
    int hh    = (id >> 3) & 15;  // h window
    int dd    = (id >> 7) & 7;   // d window
    int l     = (id >> 10) & 3;  // l within half
    int combo = (id >> 12) & 1;
    int b     = id >> 13;
    int qh  = combo;
    int kvh = 1 - combo;

    int tid = threadIdx.x;

    // Warm the shared weight table toward L2 (global_prefetch_b8).
    __builtin_prefetch(Wt + (size_t)tid * KTOT, 0, 3);

    // ---- stage x tiles for BOTH halves, bf16, K-contiguous per position ----
    {
        int c = tid;  // 128 threads = 128 input channels
        for (int h = 0; h < 2; ++h) {
            unsigned short* xt = smem_u16 + h * (16 * KTOT);
            for (int tap = 0; tap < 3; ++tap) {
                int ls = l + tap - 1;
                bool inr = (ls >= 0) && (ls < 4);   // per-half zero padding
                int Kc = tap * 128 + c;
                for (int dh = 0; dh < 4; ++dh) {
                    int di = dh >> 1, hi = dh & 1;
                    float4 val = make_float4(0.f, 0.f, 0.f, 0.f);
                    if (inr) {
                        const float* p = x
                            + ((size_t)((b * 8 + h * 4 + ls) * C_CH + c)) * DHW
                            + (2 * dd + di) * 1024 + (2 * hh + hi) * 32 + 4 * wqi;
                        val = *(const float4*)p;
                    }
                    float vv[4] = {val.x, val.y, val.z, val.w};
                    #pragma unroll
                    for (int wi = 0; wi < 4; ++wi) {
                        // pos = win*8 + di*4 + hi*2 + wi2 ; w = 4*wqi + win*2 + wi2
                        int pos = (wi >> 1) * 8 + di * 4 + hi * 2 + (wi & 1);
                        xt[pos * KTOT + Kc] = f2bf(vv[wi]);
                    }
                }
            }
        }
    }
    __syncthreads();

    // ---- WMMA GEMM: each wave computes q,k,v for 2 output-channel tiles ----
    int lane = tid & 31;
    int wid  = tid >> 5;
    int n  = lane & 15;   // N (out channel within tile) / M (position) index
    int hl = lane >> 4;   // lane half selects K sub-range / M+8
    int nt0 = wid * 2;

    v8f acc[3][2];
    const float* biases[3] = {bq, bk, bv};
    #pragma unroll
    for (int m = 0; m < 3; ++m)
        #pragma unroll
        for (int j = 0; j < 2; ++j) {
            float bv0 = biases[m][(nt0 + j) * 16 + n];
            #pragma unroll
            for (int i = 0; i < 8; ++i) acc[m][j][i] = bv0;
        }

    const unsigned short* xtQ  = smem_u16 + qh  * (16 * KTOT);
    const unsigned short* xtKV = smem_u16 + kvh * (16 * KTOT);

    for (int ks = 0; ks < 12; ++ks) {
        int kb = ks * 32;
        BF16Frag aQ, aKV;
        int ae = n * KTOT + kb + hl * 8;             // A 16x32 bf16 layout
        aQ.q[0]  = *(const uint4*)(xtQ  + ae);
        aQ.q[1]  = *(const uint4*)(xtQ  + ae + 16);
        aKV.q[0] = *(const uint4*)(xtKV + ae);
        aKV.q[1] = *(const uint4*)(xtKV + ae + 16);
        #pragma unroll
        for (int m = 0; m < 3; ++m) {
            const unsigned short* W = Wt + m * (C_CH * KTOT);
            #pragma unroll
            for (int j = 0; j < 2; ++j) {
                int o = (nt0 + j) * 16 + n;
                int be = o * KTOT + kb + hl * 8;     // B 32x16 bf16 layout
                BF16Frag bf_;
                bf_.q[0] = *(const uint4*)(W + be);
                bf_.q[1] = *(const uint4*)(W + be + 16);
                acc[m][j] = __builtin_amdgcn_wmma_f32_16x16x32_bf16(
                    false, (m == 0 ? aQ.v : aKV.v), false, bf_.v,
                    (short)0, acc[m][j], false, false);
            }
        }
    }
    __syncthreads();

    // ---- stage Q,K,V (f32) back into the same LDS: [m][pos 16][ch 128] ----
    #pragma unroll
    for (int m = 0; m < 3; ++m)
        #pragma unroll
        for (int j = 0; j < 2; ++j) {
            int ch = (nt0 + j) * 16 + n;
            #pragma unroll
            for (int r = 0; r < 8; ++r) {
                int pos = r + hl * 8;                // C/D layout: M = r (+8)
                smemF[(m * 16 + pos) * C_CH + ch] = acc[m][j][r];
            }
        }
    __syncthreads();

    // ---- window attention: 128 threads = 2 windows x 8 heads x 8 queries ---
    {
        int win  = tid >> 6;
        int head = (tid >> 3) & 7;
        int qi   = tid & 7;
        const float* Q = smemF + (win * 8 + qi) * C_CH + head * 16;
        float s[8];
        float mx = -1e30f;
        #pragma unroll
        for (int j = 0; j < 8; ++j) {
            const float* K = smemF + (16 + win * 8 + j) * C_CH + head * 16;
            float a = 0.f;
            #pragma unroll
            for (int d_ = 0; d_ < 16; ++d_) a += Q[d_] * K[d_];
            s[j] = a * 0.25f;                        // scale = 16^-0.5
            mx = fmaxf(mx, s[j]);
        }
        float sum = 0.f;
        #pragma unroll
        for (int j = 0; j < 8; ++j) { s[j] = __expf(s[j] - mx); sum += s[j]; }
        float inv = 1.f / sum;
        float out[16];
        #pragma unroll
        for (int d_ = 0; d_ < 16; ++d_) out[d_] = 0.f;
        #pragma unroll
        for (int j = 0; j < 8; ++j) {
            float p = s[j] * inv;
            const float* V = smemF + (32 + win * 8 + j) * C_CH + head * 16;
            #pragma unroll
            for (int d_ = 0; d_ < 16; ++d_) out[d_] += p * V[d_];
        }
        int di = qi >> 2, hi = (qi >> 1) & 1, wi2 = qi & 1;
        int dpos = 2 * dd + di, hpos = 2 * hh + hi, wpos = 4 * wqi + win * 2 + wi2;
        size_t base = ((size_t)((combo * 2 + b) * 4 + l) * C_CH + head * 16) * DHW
                      + dpos * 1024 + hpos * 32 + wpos;
        #pragma unroll
        for (int d_ = 0; d_ < 16; ++d_)
            Yattn[base + (size_t)d_ * DHW] = out[d_];
    }
}

// ---------------------------------------------------------------------------
// Kernel 2: projection conv4d (WMMA GEMM, M = 16 contiguous w positions).
//   Stage-in uses the CDNA5 async copy engine (global -> LDS, ASYNCcnt),
//   then an in-LDS transpose/convert builds the bf16 A-fragment buffer.
// ---------------------------------------------------------------------------
__global__ __launch_bounds__(128)
void proj_kernel(const float* __restrict__ Yattn,
                 const float* __restrict__ bp,
                 const unsigned short* __restrict__ Wtp,
                 float* __restrict__ out) {
    __shared__ __align__(16) float          stageF[KTOT * 16];  // [K][pos] 24 KB
    __shared__ __align__(16) unsigned short yT[16 * KTOT];      // [pos][K] 12 KB

    int id = blockIdx.x;
    int wt    = id & 1;           // which 16-wide w tile
    int h     = (id >> 1) & 31;
    int d     = (id >> 6) & 15;
    int l     = (id >> 10) & 3;
    int combo = (id >> 12) & 1;
    int b     = id >> 13;
    int tid = threadIdx.x;

    __builtin_prefetch(Wtp + (size_t)tid * KTOT, 0, 3);

    // ---- async copy Yattn tile into LDS (raw fp32, [K][pos] layout) --------
    {
        int c = tid;
        for (int tap = 0; tap < 3; ++tap) {
            int ls = l + tap - 1;
            float* dstrow = &stageF[(tap * 128 + c) * 16];
            if (ls >= 0 && ls < 4) {               // uniform per tap
                const float* src = Yattn
                    + ((size_t)((combo * 2 + b) * 4 + ls) * C_CH + c) * DHW
                    + d * 1024 + h * 32 + wt * 16;
                #pragma unroll
                for (int g = 0; g < 4; ++g) {
                    unsigned long long ga = (unsigned long long)(const void*)(src + 4 * g);
                    unsigned lda = (unsigned)(size_t)(const void*)(dstrow + 4 * g);
                    asm volatile("global_load_async_to_lds_b128 %0, %1, off"
                                 :: "v"(lda), "v"(ga) : "memory");
                }
            } else {
                float4 z = make_float4(0.f, 0.f, 0.f, 0.f);
                #pragma unroll
                for (int g = 0; g < 4; ++g)
                    *(float4*)(dstrow + 4 * g) = z;
            }
        }
        asm volatile("s_wait_asynccnt 0x0" ::: "memory");
    }
    __syncthreads();

    // ---- transpose + convert to bf16 A-layout [pos][K] ---------------------
    {
        #pragma unroll
        for (int t3 = 0; t3 < 3; ++t3) {
            int K = t3 * 128 + tid;
            const float* row = &stageF[K * 16];
            #pragma unroll
            for (int p = 0; p < 16; ++p)
                yT[p * KTOT + K] = f2bf(row[p]);
        }
    }
    __syncthreads();

    // ---- WMMA GEMM ---------------------------------------------------------
    int lane = tid & 31, wid = tid >> 5;
    int n = lane & 15, hl = lane >> 4;
    int nt0 = wid * 2;

    v8f acc[2];
    #pragma unroll
    for (int j = 0; j < 2; ++j) {
        float bv0 = bp[(nt0 + j) * 16 + n];
        #pragma unroll
        for (int i = 0; i < 8; ++i) acc[j][i] = bv0;
    }

    for (int ks = 0; ks < 12; ++ks) {
        int kb = ks * 32;
        BF16Frag a;
        int ae = n * KTOT + kb + hl * 8;
        a.q[0] = *(const uint4*)(yT + ae);
        a.q[1] = *(const uint4*)(yT + ae + 16);
        #pragma unroll
        for (int j = 0; j < 2; ++j) {
            int o = (nt0 + j) * 16 + n;
            int be = o * KTOT + kb + hl * 8;
            BF16Frag bf_;
            bf_.q[0] = *(const uint4*)(Wtp + be);
            bf_.q[1] = *(const uint4*)(Wtp + be + 16);
            acc[j] = __builtin_amdgcn_wmma_f32_16x16x32_bf16(
                false, a.v, false, bf_.v, (short)0, acc[j], false, false);
        }
    }

    #pragma unroll
    for (int j = 0; j < 2; ++j) {
        int o = (nt0 + j) * 16 + n;
        #pragma unroll
        for (int r = 0; r < 8; ++r) {
            int pos = r + hl * 8;
            size_t oidx = ((size_t)((b * 8 + combo * 4 + l) * C_CH + o)) * DHW
                        + d * 1024 + h * 32 + wt * 16 + pos;
            out[oidx] = acc[j][r];
        }
    }
}

// ---------------------------------------------------------------------------
extern "C" void kernel_launch(void* const* d_in, const int* in_sizes, int n_in,
                              void* d_out, int out_size, void* d_ws, size_t ws_size,
                              hipStream_t stream) {
    const float* x  = (const float*)d_in[0];
    const float* wq = (const float*)d_in[1];
    const float* bq = (const float*)d_in[2];
    const float* wk = (const float*)d_in[3];
    const float* bk = (const float*)d_in[4];
    const float* wv = (const float*)d_in[5];
    const float* bv = (const float*)d_in[6];
    const float* wp = (const float*)d_in[7];
    const float* bp = (const float*)d_in[8];

    // workspace layout: [4 * 128*384 bf16 transposed weights][Yattn f32 128MiB]
    unsigned short* Wt = (unsigned short*)d_ws;
    float* Yattn = (float*)((char*)d_ws + (size_t)4 * C_CH * KTOT * sizeof(unsigned short));

    wt_transpose_kernel<<<768, 256, 0, stream>>>(wq, wk, wv, wp, Wt);
    qkv_attn_kernel<<<16384, 128, 0, stream>>>(x, bq, bk, bv, Wt, Yattn);
    proj_kernel<<<16384, 128, 0, stream>>>(Yattn, bp, Wt + (size_t)3 * C_CH * KTOT,
                                           (float*)d_out);
}
